// BaseTFMBlock_45930380263445
// MI455X (gfx1250) — compile-verified, matrix operand order checked by reading, not run on hardware
//
#include <hip/hip_runtime.h>

// ---------------------------------------------------------------------------
// Fused transformer block for MI455X (gfx1250, wave32, WMMA 16x16x32 bf16)
// B=4, S=1024, H=1024, NH=16, DK=64
// ---------------------------------------------------------------------------

typedef __attribute__((ext_vector_type(8)))  __bf16 v8bf;
typedef __attribute__((ext_vector_type(16))) __bf16 v16bf;
typedef __attribute__((ext_vector_type(8)))  float  v8f;

static constexpr int Bb = 4;
static constexpr int S  = 1024;
static constexpr int H  = 1024;
static constexpr int NH = 16;
static constexpr int DK = 64;
static constexpr int MTOK = Bb * S;   // 4096 tokens

__device__ __forceinline__ v8f wmma_bf16(v16bf a, v16bf b, v8f c) {
  return __builtin_amdgcn_wmma_f32_16x16x32_bf16(
      /*neg_a=*/false, a, /*neg_b=*/false, b,
      /*c_mod=*/(short)0, c, /*reuse_a=*/false, /*reuse_b=*/false);
}

// Load a 16x32 bf16 A/B fragment half-row for this lane.
// Per ISA 7.12.2: lane holds K = 8*(lane>>4)+{0..7} and 16+8*(lane>>4)+{0..7}.
// p must point at (row_base + 8*(lane>>4)); both 16B chunks are aligned.
__device__ __forceinline__ v16bf load_frag(const __bf16* p) {
  v8bf lo = *(const v8bf*)(p);
  v8bf hi = *(const v8bf*)(p + 16);
  return __builtin_shufflevector(lo, hi, 0,1,2,3,4,5,6,7,8,9,10,11,12,13,14,15);
}

// ---- DPP16 butterfly reductions over the 16-lane half-row (pure VALU, no LDS)
template <int CTRL>
__device__ __forceinline__ float dpp_mov(float x) {
  int r = __builtin_amdgcn_update_dpp(0, __builtin_bit_cast(int, x),
                                      CTRL, 0xf, 0xf, true);
  return __builtin_bit_cast(float, r);
}
__device__ __forceinline__ float red_max16(float x) {
  x = fmaxf(x, dpp_mov<0xB1>(x));    // quad_perm [1,0,3,2]  : xor 1
  x = fmaxf(x, dpp_mov<0x4E>(x));    // quad_perm [2,3,0,1]  : xor 2
  x = fmaxf(x, dpp_mov<0x141>(x));   // row_half_mirror      : combine quads in 8
  x = fmaxf(x, dpp_mov<0x140>(x));   // row_mirror           : combine 8-halves in 16
  return x;
}
__device__ __forceinline__ float red_sum16(float x) {
  x += dpp_mov<0xB1>(x);
  x += dpp_mov<0x4E>(x);
  x += dpp_mov<0x141>(x);
  x += dpp_mov<0x140>(x);
  return x;
}

// ---------------------------------------------------------------------------
// fp32 -> bf16 conversion (grid-stride, non-temporal reads: streamed once)
// ---------------------------------------------------------------------------
__global__ void cvt_f32_bf16(const float* __restrict__ in,
                             __bf16* __restrict__ out, int n) {
  int i = blockIdx.x * blockDim.x + threadIdx.x;
  int stride = gridDim.x * blockDim.x;
  for (; i < n; i += stride) out[i] = (__bf16)__builtin_nontemporal_load(in + i);
}

// ---------------------------------------------------------------------------
// GEMM: Y = X (4096x1024) @ W^T (1024x1024), bf16 in, WMMA f32 accum.
// Block = 4 waves sharing one 64-wide N tile; the shared 64x64 W tile is
// async-staged into LDS (ASYNCcnt) and double-buffered. Each wave owns a
// 32x64 output tile: 2x4 accumulators, 16 WMMAs per barrier (two k=32 steps).
// MODE 0: store bf16 row-major [m][n]
// MODE 1: store bf16 transposed per-head: VT[(b*NH+h)*DK + dk][s]
// MODE 2: store f32 = acc + resid[m][n]   (final output projection)
// ---------------------------------------------------------------------------
static constexpr int WPAD = 72;  // padded W-tile row stride (144B, 16B-aligned)

template <int MODE>
__global__ __launch_bounds__(128)
void gemm_bf16(const __bf16* __restrict__ X, const __bf16* __restrict__ W,
               void* __restrict__ out, const float* __restrict__ resid) {
  constexpr int Kdim = H;
  constexpr int Ndim = H;

  __shared__ __bf16 Bst[2][64][WPAD];   // 64 N-rows x 64 K (padded), x2 buffers

  const int tid  = threadIdx.x;
  const int lane = tid & 31;
  const int wv   = tid >> 5;
  const int g    = lane >> 4;     // lo/hi 16-lane half
  const int r16  = lane & 15;

  const int tileN = blockIdx.x * 64;                 // shared by all 4 waves
  const int tileM = (blockIdx.y * 4 + wv) * 32;

  // async-stage a 64x64 bf16 W tile: 512 16B chunks over 128 threads
  auto issue_w = [&](int kb, int buf) {
#pragma unroll
    for (int c = 0; c < 4; ++c) {
      const int ch  = tid * 4 + c;          // 0..511
      const int row = ch >> 3, col = (ch & 7) * 8;
      const __bf16* src = W + (size_t)(tileN + row) * Kdim + kb + col;
      unsigned dst = (unsigned)(size_t)&Bst[buf][row][col];
      asm volatile("global_load_async_to_lds_b128 %0, %1, off"
                   :: "v"(dst), "v"(src) : "memory");
    }
  };

  v8f acc[2][4] = {};

  const __bf16* Arow[2];
#pragma unroll
  for (int r = 0; r < 2; ++r)
    Arow[r] = X + (size_t)(tileM + r * 16 + r16) * Kdim + 8 * g;

  issue_w(0, 0);
  asm volatile("s_wait_asynccnt 0x0" ::: "memory");
  __syncthreads();

  for (int kb = 0; kb < Kdim; kb += 64) {
    const int buf = (kb >> 6) & 1;
    if (kb + 64 < Kdim) issue_w(kb + 64, buf ^ 1);   // prefetch next W tile

#pragma unroll
    for (int kk = 0; kk < 64; kk += 32) {
      v16bf a[2], b[4];
#pragma unroll
      for (int r = 0; r < 2; ++r) a[r] = load_frag(Arow[r] + kb + kk);
#pragma unroll
      for (int c = 0; c < 4; ++c)
        b[c] = load_frag(&Bst[buf][c * 16 + r16][kk + 8 * g]);
#pragma unroll
      for (int r = 0; r < 2; ++r)
#pragma unroll
        for (int c = 0; c < 4; ++c) acc[r][c] = wmma_bf16(a[r], b[c], acc[r][c]);
    }

    asm volatile("s_wait_asynccnt 0x0" ::: "memory");
    __syncthreads();
  }

  // Epilogue. C/D layout: lane holds column n = sub*16 + r16,
  // rows m = m0..m0+7 with m0 = tile + r*16 + 8*g (one row per VGPR).
#pragma unroll
  for (int r = 0; r < 2; ++r) {
#pragma unroll
    for (int c = 0; c < 4; ++c) {
      const int n  = tileN + c * 16 + r16;
      const int m0 = tileM + r * 16 + 8 * g;
      if constexpr (MODE == 0) {
        __bf16* O = (__bf16*)out;
#pragma unroll
        for (int v = 0; v < 8; ++v)
          O[(size_t)(m0 + v) * Ndim + n] = (__bf16)acc[r][c][v];
      } else if constexpr (MODE == 1) {
        __bf16* O = (__bf16*)out;
        const int hh = n >> 6, dk = n & 63;
        const int bb = m0 >> 10, s0 = m0 & 1023;  // 8 consecutive s, same bb
        v8bf pk;
#pragma unroll
        for (int v = 0; v < 8; ++v) pk[v] = (__bf16)acc[r][c][v];
        *(v8bf*)(O + ((size_t)(bb * NH + hh) * DK + dk) * S + s0) = pk;
      } else {
        float* O = (float*)out;
#pragma unroll
        for (int v = 0; v < 8; ++v) {
          const size_t idx = (size_t)(m0 + v) * Ndim + n;
          O[idx] = acc[r][c][v] + resid[idx];
        }
      }
    }
  }
}

// ---------------------------------------------------------------------------
// Flash attention. Block = 4 waves, all on the same (b,h) -> K/V tiles are
// staged ONCE per block into LDS with gfx1250 async-to-LDS copies
// (ASYNCcnt), double-buffered against the WMMA compute. Per wave, one
// 16-query tile; 32 keys per iteration.
// ---------------------------------------------------------------------------
static constexpr int KP = 72;   // padded K-tile row stride (bf16 elems, 144B)
static constexpr int VP = 40;   // padded V-tile row stride (bf16 elems,  80B)

__global__ __launch_bounds__(128)
void attn_kernel(const __bf16* __restrict__ Qb, const __bf16* __restrict__ Kb,
                 const __bf16* __restrict__ VTb, const float* __restrict__ bias,
                 __bf16* __restrict__ Ob) {
  __shared__ __bf16 Kst[2][32][KP];   // 32 keys x 64 dk   (padded)
  __shared__ __bf16 Vst[2][64][VP];   // 64 dk   x 32 keys (padded)
  __shared__ __bf16 plds[4][16 * 32]; // per-wave P staging

  const int tid  = threadIdx.x;
  const int lane = tid & 31;
  const int wv   = tid >> 5;
  const int g    = lane >> 4;
  const int r16  = lane & 15;

  const int bh = blockIdx.y;            // 0..63
  const int b  = bh >> 4;
  const int h  = bh & 15;
  const int q0 = (blockIdx.x * 4 + wv) * 16;

  const size_t tokBase = (size_t)b * S;
  const float*  biasBH = bias + (size_t)bh * S * S;
  const __bf16* Kbh    = Kb + tokBase * H + h * DK;   // + key*H
  const __bf16* VTbh   = VTb + (size_t)bh * DK * S;   // + dk*S

  // async-stage one 32-key K tile + V^T tile into LDS buffer `buf`
  auto issue_tile = [&](int kb, int buf) {
#pragma unroll
    for (int c = 0; c < 2; ++c) {           // K: 256 16B chunks / 128 thr
      const int ch  = tid * 2 + c;
      const int key = ch >> 3, col = (ch & 7) * 8;
      const __bf16* src = Kbh + (size_t)(kb + key) * H + col;
      unsigned dst = (unsigned)(size_t)&Kst[buf][key][col];
      asm volatile("global_load_async_to_lds_b128 %0, %1, off"
                   :: "v"(dst), "v"(src) : "memory");
    }
#pragma unroll
    for (int c = 0; c < 2; ++c) {           // V: 256 16B chunks / 128 thr
      const int ch = tid * 2 + c;
      const int dk = ch >> 2, col = (ch & 3) * 8;
      const __bf16* src = VTbh + (size_t)dk * S + kb + col;
      unsigned dst = (unsigned)(size_t)&Vst[buf][dk][col];
      asm volatile("global_load_async_to_lds_b128 %0, %1, off"
                   :: "v"(dst), "v"(src) : "memory");
    }
  };

  // Q fragments for both dk halves (persist across the whole key loop)
  const __bf16* qrow = Qb + (tokBase + q0 + r16) * H + h * DK + 8 * g;
  const v16bf aq0 = load_frag(qrow);
  const v16bf aq1 = load_frag(qrow + 32);

  float Mi[8], Li[8];
  v8f oacc[4] = {};
#pragma unroll
  for (int v = 0; v < 8; ++v) { Mi[v] = -1e30f; Li[v] = 0.f; }

  __bf16* pw = &plds[wv][0];

  issue_tile(0, 0);
  asm volatile("s_wait_asynccnt 0x0" ::: "memory");
  __syncthreads();

  for (int kb = 0; kb < S; kb += 32) {
    const int buf = (kb >> 5) & 1;
    if (kb + 32 < S) issue_tile(kb + 32, buf ^ 1);   // prefetch next tile

    // scores: two 16x16 key tiles from the LDS-staged K tile
    v8f s0, s1;
    {
      const __bf16* krow = &Kst[buf][r16][8 * g];
      v8f z = {};
      z  = wmma_bf16(aq0, load_frag(krow), z);
      s0 = wmma_bf16(aq1, load_frag(krow + 32), z);
    }
    {
      const __bf16* krow = &Kst[buf][16 + r16][8 * g];
      v8f z = {};
      z  = wmma_bf16(aq0, load_frag(krow), z);
      s1 = wmma_bf16(aq1, load_frag(krow + 32), z);
    }

    // bias (non-temporal: 256MB streamed once) + scale + online softmax.
    // Row = fixed VGPR, spread over 16 lanes -> DPP butterfly reductions.
    float p0[8], p1[8], corr[8];
#pragma unroll
    for (int v = 0; v < 8; ++v) {
      const int mrow = q0 + 8 * g + v;
      const float b0 =
          __builtin_nontemporal_load(biasBH + (size_t)mrow * S + kb + r16);
      const float b1 =
          __builtin_nontemporal_load(biasBH + (size_t)mrow * S + kb + 16 + r16);
      float t0 = s0[v] * 0.125f + b0;
      float t1 = s1[v] * 0.125f + b1;
      const float rm = red_max16(fmaxf(t0, t1));
      const float nm = fmaxf(Mi[v], rm);
      const float cf = __expf(Mi[v] - nm);
      const float e0 = __expf(t0 - nm);
      const float e1 = __expf(t1 - nm);
      const float rs = red_sum16(e0 + e1);
      Li[v] = Li[v] * cf + rs;
      Mi[v] = nm;
      p0[v] = e0; p1[v] = e1; corr[v] = cf;
    }
#pragma unroll
    for (int t = 0; t < 4; ++t)
#pragma unroll
      for (int v = 0; v < 8; ++v) oacc[t][v] *= corr[v];

    // P: D-layout (lane=col) -> LDS row-major [m][k] -> A-layout reload.
    // Same-wave LDS ops are in-order on CDNA5 (DScnt), no barrier needed.
#pragma unroll
    for (int v = 0; v < 8; ++v) {
      pw[(8 * g + v) * 32 + r16]      = (__bf16)p0[v];
      pw[(8 * g + v) * 32 + 16 + r16] = (__bf16)p1[v];
    }
    __builtin_amdgcn_wave_barrier();
    union { v16bf v; v8bf h2[2]; } pu;
    pu.h2[0] = *(const v8bf*)(pw + r16 * 32 + 8 * g);
    pu.h2[1] = *(const v8bf*)(pw + r16 * 32 + 16 + 8 * g);

    // P @ V from the LDS-staged V^T tile (contiguous along keys)
#pragma unroll
    for (int t = 0; t < 4; ++t) {
      const __bf16* vrow = &Vst[buf][t * 16 + r16][8 * g];
      oacc[t] = wmma_bf16(pu.v, load_frag(vrow), oacc[t]);
    }

    // next-tile async copies must land + all waves done reading this buffer
    asm volatile("s_wait_asynccnt 0x0" ::: "memory");
    __syncthreads();
  }

  // normalize + store attention output rows into [token][h*DK+dk] bf16
#pragma unroll
  for (int t = 0; t < 4; ++t) {
    const int dk = t * 16 + r16;
#pragma unroll
    for (int v = 0; v < 8; ++v) {
      const int mrow = q0 + 8 * g + v;
      Ob[(tokBase + mrow) * H + h * DK + dk] = (__bf16)(oacc[t][v] / Li[v]);
    }
  }
}

// ---------------------------------------------------------------------------
// Host-side launcher
// ---------------------------------------------------------------------------
extern "C" void kernel_launch(void* const* d_in, const int* in_sizes, int n_in,
                              void* d_out, int out_size, void* d_ws, size_t ws_size,
                              hipStream_t stream) {
  const float* x_q  = (const float*)d_in[0];
  const float* x_kv = (const float*)d_in[1];
  const float* bias = (const float*)d_in[2];
  const float* Wq   = (const float*)d_in[3];
  const float* Wk   = (const float*)d_in[4];
  const float* Wv   = (const float*)d_in[5];
  const float* Wo   = (const float*)d_in[6];
  float* out = (float*)d_out;

  char* ws = (char*)d_ws;
  const size_t MB = 1024ull * 1024ull;
  __bf16* xq_bf  = (__bf16*)(ws + 0 * MB);    // 8 MB
  __bf16* xkv_bf = (__bf16*)(ws + 8 * MB);    // 8 MB
  __bf16* wq_bf  = (__bf16*)(ws + 16 * MB);   // 2 MB each
  __bf16* wk_bf  = (__bf16*)(ws + 18 * MB);
  __bf16* wv_bf  = (__bf16*)(ws + 20 * MB);
  __bf16* wo_bf  = (__bf16*)(ws + 22 * MB);
  __bf16* Qb     = (__bf16*)(ws + 24 * MB);   // 8 MB
  __bf16* Kb     = (__bf16*)(ws + 32 * MB);   // 8 MB
  __bf16* VTb    = (__bf16*)(ws + 40 * MB);   // 8 MB (per-head transposed)
  __bf16* Ob     = (__bf16*)(ws + 48 * MB);   // 8 MB

  const int NX = MTOK * H;   // 4M elements
  const int NW = H * H;      // 1M elements
  cvt_f32_bf16<<<1024, 256, 0, stream>>>(x_q,  xq_bf,  NX);
  cvt_f32_bf16<<<1024, 256, 0, stream>>>(x_kv, xkv_bf, NX);
  cvt_f32_bf16<<<512,  256, 0, stream>>>(Wq, wq_bf, NW);
  cvt_f32_bf16<<<512,  256, 0, stream>>>(Wk, wk_bf, NW);
  cvt_f32_bf16<<<512,  256, 0, stream>>>(Wv, wv_bf, NW);
  cvt_f32_bf16<<<512,  256, 0, stream>>>(Wo, wo_bf, NW);

  // 4096x1024x1024 GEMMs: wave tile 32x64, grid = (N/64, M/32/4waves)
  dim3 ggrid(H / 64, MTOK / 32 / 4), gblk(128);
  gemm_bf16<0><<<ggrid, gblk, 0, stream>>>(xq_bf,  wq_bf, (void*)Qb,  nullptr);
  gemm_bf16<0><<<ggrid, gblk, 0, stream>>>(xkv_bf, wk_bf, (void*)Kb,  nullptr);
  gemm_bf16<1><<<ggrid, gblk, 0, stream>>>(xkv_bf, wv_bf, (void*)VTb, nullptr);

  // attention: (S/16 tiles / 4 waves, B*NH)
  attn_kernel<<<dim3(S / 16 / 4, Bb * NH), 128, 0, stream>>>(Qb, Kb, VTb, bias, Ob);

  // output projection + residual, f32 out
  gemm_bf16<2><<<ggrid, gblk, 0, stream>>>(Ob, wo_bf, (void*)out, x_q);
}
